// BSHConv3D_29661044146298
// MI455X (gfx1250) — matrix operand order repaired and checked by reading, not compiled, not run on hardware
//
#include <hip/hip_runtime.h>
#include <hip/hip_bf16.h>

// ---- problem constants (fixed by the reference) ----
#define DD   48
#define HH   48
#define WW   48
#define CIN  8
#define NOUT 512                 // 16 filters * 16 harmonics * 2 (re/im), GEMM N
#define KRAW 216                 // 27 taps * 8 channels, GEMM K
#define KPAD 224                 // padded to 7 * 32 (B matrix row stride)
#define LDA  232                 // LDS A-tile row stride (bank-conflict-free pad)
#define MTOT (DD*HH*WW)          // 110592 voxels, GEMM M
#define MTILE 64                 // voxels per block
#define NTAP 27

typedef __attribute__((ext_vector_type(16))) _Float16 v16h;
typedef __attribute__((ext_vector_type(8)))  _Float16 v8h;
typedef __attribute__((ext_vector_type(8)))  float    v8f;

// ---------------------------------------------------------------------------
// Prep: fold atoms_{real,imag} [3,3,3,2,16], w [8,16,2,4] (degree-expanded),
// and w_center [8,16] into a single K-major f16 weight matrix Wt[512][224],
// plus an f32 bias[512] (b_center on channels n%32==0).
// outchan n = f*32 + nh*2 + reim   (matches output layout [f, nh, re/im])
// k       = tap*8 + c, tap = ((dz*3)+dy)*3+dx, center tap = 13
// ---------------------------------------------------------------------------
__global__ void bsh_prep_weights(const float* __restrict__ atoms_r,
                                 const float* __restrict__ atoms_i,
                                 const float* __restrict__ w,
                                 const float* __restrict__ wc,
                                 const float* __restrict__ bc,
                                 _Float16* __restrict__ Wt,
                                 float* __restrict__ bias) {
    int g = blockIdx.x * 256 + threadIdx.x;
    if (g < NOUT) {
        bias[g] = ((g & 31) == 0) ? bc[g >> 5] : 0.0f;
    }
    if (g >= NOUT * KPAD) return;
    int n = g / KPAD;
    int k = g - n * KPAD;
    int f    = n >> 5;
    int nh   = (n >> 1) & 15;
    int reim = n & 1;
    float val = 0.0f;
    if (k < KRAW) {
        int c   = k & 7;
        int tap = k >> 3;
        int d = (nh == 0) ? 0 : (nh < 4) ? 1 : (nh < 9) ? 2 : 3;  // degree of harmonic
        const float* atoms = reim ? atoms_i : atoms_r;
        #pragma unroll
        for (int r = 0; r < 2; ++r)
            val += atoms[(tap * 2 + r) * 16 + nh] * w[((c * 16 + f) * 2 + r) * 4 + d];
        if (reim == 0 && nh == 0 && tap == 13)   // central 1x1x1 conv into real, n=0
            val += wc[c * 16 + f];
    }
    Wt[n * KPAD + k] = (_Float16)val;
}

// ---------------------------------------------------------------------------
// Main GEMM: out[M=110592][N=512] = im2col(x)[M][K=216] * Wt^T + bias
// Block: 512 threads (16 waves), MTILE=64 voxels, full N=512.
// Wave wv owns N slice [wv*32, wv*32+32) across all 64 M rows:
//   7 K-chunks x (4 M-subtiles x 2 N-subtiles) = 56 v_wmma_f32_16x16x32_f16,
//   B fragments loaded once per (chunk, N-subtile), reused over 4 M-subtiles.
// Each block reads the 229 KB weight matrix exactly once -> ~0.4 GB total
// L2 traffic for the whole dispatch (vs 1.58 GB with MTILE=16).
// ---------------------------------------------------------------------------
__global__ void __launch_bounds__(512)
bsh_conv_gemm(const float* __restrict__ x,
              const _Float16* __restrict__ Wt,
              const float* __restrict__ bias,
              float* __restrict__ outp) {
    __shared__ _Float16 Atile[MTILE * LDA];   // 29696 bytes

    const int tid = threadIdx.x;
    const int p0  = blockIdx.x * MTILE;       // first voxel of this tile

    // ---- cooperative vectorized im2col into LDS ----
    // task = (m, tapg): tapg<27 -> 8 input channels of one neighbor voxel
    //                   tapg==27 -> zero-fill K padding [216,224)
    for (int t = tid; t < MTILE * (NTAP + 1); t += 512) {
        int m    = t / (NTAP + 1);
        int tapg = t - m * (NTAP + 1);
        v8h vals = {};
        if (tapg < NTAP) {
            int p  = p0 + m;
            int zz = p / (HH * WW)  + tapg / 9 - 1;
            int yy = (p / WW) % HH  + (tapg / 3) % 3 - 1;
            int xx = p % WW         + tapg % 3 - 1;
            if ((unsigned)xx < (unsigned)WW && (unsigned)yy < (unsigned)HH &&
                (unsigned)zz < (unsigned)DD) {
                const float* src = &x[(((zz * HH) + yy) * WW + xx) * CIN];
                #pragma unroll
                for (int c = 0; c < CIN; ++c) vals[c] = (_Float16)src[c];
            }
        }
        *(v8h*)&Atile[m * LDA + tapg * 8] = vals;   // 16B-aligned ds_store_b128
    }

    const int lane  = tid & 31;
    const int wv    = tid >> 5;       // wave id 0..15 -> N base wv*32
    const int h     = lane >> 4;      // lane half
    const int mrow  = lane & 15;      // A fragment row within M-subtile
    const int nlane = lane & 15;      // B/C fragment column

    // warm L2 with this wave's weight slice (global_prefetch_b8)
    __builtin_prefetch(&Wt[(wv * 32 + nlane) * KPAD], 0, 1);

    __syncthreads();

    v8f acc[4][2] = {};

    #pragma unroll
    for (int kc = 0; kc < 7; ++kc) {
        const int k0 = kc * 32;

        // A fragments: VGPRs 0-3 hold K = h*16..h*16+7,
        // VGPRs 4-7 hold K = h*16+8..h*16+15, row M = mi*16 + mrow.
        v16h afrag[4];
        #pragma unroll
        for (int mi = 0; mi < 4; ++mi) {
            const int row = mi * 16 + mrow;
            const v8h alo = *(const v8h*)&Atile[row * LDA + k0 + h * 16];
            const v8h ahi = *(const v8h*)&Atile[row * LDA + k0 + h * 16 + 8];
            afrag[mi] = __builtin_shufflevector(alo, ahi,
                0, 1, 2, 3, 4, 5, 6, 7, 8, 9, 10, 11, 12, 13, 14, 15);
        }

        // B fragments: lane holds K = k0+h*16 .. +15 of column n (K-major Wt)
        #pragma unroll
        for (int s = 0; s < 2; ++s) {
            const int n = wv * 32 + s * 16 + nlane;
            const v16h bfrag = *(const v16h*)&Wt[n * KPAD + k0 + h * 16];
            #pragma unroll
            for (int mi = 0; mi < 4; ++mi) {
                acc[mi][s] = __builtin_amdgcn_wmma_f32_16x16x32_f16(
                    /*neg_a=*/false, afrag[mi], /*neg_b=*/false, bfrag,
                    /*c_mod=*/(short)0, acc[mi][s],
                    /*reuse_a=*/false, /*reuse_b=*/false);
            }
        }
    }

    // ---- epilogue: C/D layout VGPR v -> row M = v + 8*h, col N = nlane ----
    #pragma unroll
    for (int s = 0; s < 2; ++s) {
        const int n  = wv * 32 + s * 16 + nlane;
        const float bv = bias[n];
        #pragma unroll
        for (int mi = 0; mi < 4; ++mi) {
            const size_t base = ((size_t)p0 + (size_t)mi * 16) * NOUT + (size_t)n;
            #pragma unroll
            for (int v = 0; v < 8; ++v) {
                outp[base + (size_t)(v + 8 * h) * NOUT] = acc[mi][s][v] + bv;
            }
        }
    }
}

// ---------------------------------------------------------------------------
// inputs (setup_inputs order): x, atoms_real, atoms_imag, w, w_center, b_center
// output: f32 [48,48,48,16,16,2] = 56,623,104 floats
// workspace: Wt f16 512*224 = 229376 B  @ offset 0
//            bias f32 512   =   2048 B  @ offset 229376
// ---------------------------------------------------------------------------
extern "C" void kernel_launch(void* const* d_in, const int* in_sizes, int n_in,
                              void* d_out, int out_size, void* d_ws, size_t ws_size,
                              hipStream_t stream) {
    const float* x       = (const float*)d_in[0];
    const float* atoms_r = (const float*)d_in[1];
    const float* atoms_i = (const float*)d_in[2];
    const float* w       = (const float*)d_in[3];
    const float* wc      = (const float*)d_in[4];
    const float* bc      = (const float*)d_in[5];

    _Float16* Wt   = (_Float16*)d_ws;
    float*    bias = (float*)((char*)d_ws + (size_t)NOUT * KPAD * sizeof(_Float16));
    float*    outp = (float*)d_out;

    const int prep_threads = NOUT * KPAD;                       // 114688
    bsh_prep_weights<<<(prep_threads + 255) / 256, 256, 0, stream>>>(
        atoms_r, atoms_i, w, wc, bc, Wt, bias);

    bsh_conv_gemm<<<MTOT / MTILE, 512, 0, stream>>>(x, Wt, bias, outp);
}
